// MultiHeadSelfAttention_56298431316301
// MI455X (gfx1250) — matrix operand order repaired and checked by reading, not compiled, not run on hardware
//
#include <hip/hip_runtime.h>
#include <hip/hip_bf16.h>

// ---------------------------------------------------------------------------
// MI455X (gfx1250) multi-head self-attention forward.
// B=4, S=1024, H=16, DK=DV=64, D_MODEL=1024. fp32 in/out, bf16 WMMA compute
// (v_wmma_f32_16x16x32_bf16) with fp32 accumulation, fp32 softmax.
//
// All WMMA operands are laid out K-contiguous so every fragment load is a
// 16/32-byte vector load. X is staged to LDS with double-buffered
// global_load_async_to_lds_b128 (ASYNCcnt). attn (268MB, never re-read) is
// written once with nontemporal stores so L2 (192MB) keeps Q/K/V/W resident.
// ---------------------------------------------------------------------------

typedef __bf16 bf16;
typedef __attribute__((ext_vector_type(16))) __bf16 v16bf;
typedef __attribute__((ext_vector_type(8)))  __bf16 v8bf;
typedef __attribute__((ext_vector_type(8)))  float  v8f;
typedef __attribute__((ext_vector_type(4)))  float  v4f;

#define D_MODEL 1024
#define S_LEN   1024
#define NH      16
#define DK      64

__device__ __forceinline__ v8f wmma_bf16(v16bf a, v16bf b, v8f c) {
  return __builtin_amdgcn_wmma_f32_16x16x32_bf16(
      false, a, false, b, (short)0, c, false, false);
}

__device__ __forceinline__ v16bf cat8(v8bf lo, v8bf hi) {
  return __builtin_shufflevector(lo, hi, 0, 1, 2, 3, 4, 5, 6, 7,
                                 8, 9, 10, 11, 12, 13, 14, 15);
}

// A-frag 16x32 bf16, K contiguous, row stride rs (elements).
// lanes 0-15: row=lane, elems 0..7=K kh..kh+7, 8..15=K 16+kh.. (kh=0)
// lanes 16-31: kh=8.
__device__ __forceinline__ v16bf frag_a16(const bf16* p, int rs) {
  const int lane = threadIdx.x & 31;
  const bf16* q = p + (size_t)(lane & 15) * rs + ((lane >> 4) << 3);
  v8bf lo = *(const v8bf*)q;
  v8bf hi = *(const v8bf*)(q + 16);
  return cat8(lo, hi);
}

// A-frag from fp32 source (global or LDS), K contiguous, convert to bf16.
__device__ __forceinline__ v16bf frag_a32(const float* p, int rs) {
  const int lane = threadIdx.x & 31;
  const float* q = p + (size_t)(lane & 15) * rs + ((lane >> 4) << 3);
  v4f a0 = *(const v4f*)q;
  v4f a1 = *(const v4f*)(q + 4);
  v4f b0 = *(const v4f*)(q + 16);
  v4f b1 = *(const v4f*)(q + 20);
  v16bf r;
#pragma unroll
  for (int i = 0; i < 4; ++i) {
    r[i]      = (bf16)a0[i];
    r[4 + i]  = (bf16)a1[i];
    r[8 + i]  = (bf16)b0[i];
    r[12 + i] = (bf16)b1[i];
  }
  return r;
}

// B-frag 32x16 bf16, K contiguous per column, column stride cs (elements).
// lanes 0-15: col=lane, elems 0..7=K kh..kh+7, 8..15=K 16+kh.. (kh=0); lanes 16-31: kh=8.
__device__ __forceinline__ v16bf frag_b16(const bf16* p, int cs) {
  const int lane = threadIdx.x & 31;
  const bf16* q = p + (size_t)(lane & 15) * cs + ((lane >> 4) << 3);
  v8bf lo = *(const v8bf*)q;
  v8bf hi = *(const v8bf*)(q + 16);
  return cat8(lo, hi);
}

// ---- gfx1250 async DMA: global -> LDS, 16B per lane -----------------------
__device__ __forceinline__ unsigned lds_off(const void* p) {
  return (unsigned)(unsigned long long)p;   // generic LDS addr low 32 = offset
}
__device__ __forceinline__ void async_b128(unsigned dst_lds, const void* src) {
  asm volatile("global_load_async_to_lds_b128 %0, %1, off"
               :: "v"(dst_lds), "v"(src) : "memory");
}
#define WAIT_ASYNC(n) asm volatile("s_wait_asynccnt " #n ::: "memory")

// ---------------------------------------------------------------------------
// Kernel 0: one-time transpose+convert: src [batch][R][C] f32 -> dst [batch][C][R] bf16
// ---------------------------------------------------------------------------
__global__ __launch_bounds__(256)
void tcvt_kernel(const float* __restrict__ src, bf16* __restrict__ dst,
                 int R, int C) {
  __shared__ float t[64][65];
  const int batch = blockIdx.x;
  const int r0 = blockIdx.y * 64;
  const int c0 = blockIdx.z * 64;
  const float* s = src + ((size_t)batch * R + r0) * C + c0;
  bf16* d = dst + ((size_t)batch * C + c0) * R + r0;
  const int tx = threadIdx.x & 63;
  const int ty = threadIdx.x >> 6;
#pragma unroll
  for (int i = 0; i < 16; ++i) {
    const int r = ty * 16 + i;
    t[r][tx] = s[(size_t)r * C + tx];
  }
  __syncthreads();
#pragma unroll
  for (int i = 0; i < 16; ++i) {
    const int c = ty * 16 + i;
    d[(size_t)c * R + tx] = (bf16)t[tx][c];
  }
}

// ---------------------------------------------------------------------------
// Kernel 1: Q/K/V projections. grid(B*H, S/64), 8 waves.
// X chunk (64x32 f32, 8KB) double-buffered in LDS via async DMA; weight
// fragments pulled K-contiguous from pre-transposed bf16 WT (L2-resident).
// Q,K stored [B,H,S,64]; V stored transposed [B,H,64,S].
// ---------------------------------------------------------------------------
__global__ __launch_bounds__(256)
void qkv_kernel(const float* __restrict__ x,
                const bf16* __restrict__ WqT, const float* __restrict__ bq,
                const bf16* __restrict__ WkT, const float* __restrict__ bk,
                const bf16* __restrict__ WvT, const float* __restrict__ bv,
                bf16* __restrict__ Q, bf16* __restrict__ K, bf16* __restrict__ VT) {
  __shared__ float s_x[2][64 * 32];     // 16 KB double buffer

  const int bh = blockIdx.x;
  const int b  = bh >> 4;
  const int h  = bh & 15;
  const int s0 = blockIdx.y * 64;
  const int wave = threadIdx.x >> 5;
  const int mt = wave & 3;
  const int g  = wave >> 2;
  const int tid = threadIdx.x;

  const float* xb = x + ((size_t)(b * S_LEN + s0)) * D_MODEL;

  // stage chunk 0
#pragma unroll
  for (int it = 0; it < 2; ++it) {
    const int f = (tid + it * 256) * 4;            // flat float idx in 64x32 tile
    async_b128(lds_off(&s_x[0][f]), xb + (size_t)(f >> 5) * D_MODEL + (f & 31));
  }

  v8f acc[6];
#pragma unroll
  for (int j = 0; j < 6; ++j) acc[j] = (v8f){};

  for (int kc = 0; kc < 32; ++kc) {
    const float* cur = &s_x[kc & 1][0];
    if (kc + 1 < 32) {
      const int k0n = (kc + 1) * 32;
#pragma unroll
      for (int it = 0; it < 2; ++it) {
        const int f = (tid + it * 256) * 4;
        async_b128(lds_off(&s_x[(kc + 1) & 1][f]),
                   xb + (size_t)(f >> 5) * D_MODEL + k0n + (f & 31));
      }
      WAIT_ASYNC(0x2);                 // in-order: previous chunk complete
    } else {
      WAIT_ASYNC(0x0);
    }
    __syncthreads();

    const int k0 = kc * 32;
    v16bf a = frag_a32(cur + mt * 16 * 32, 32);
#pragma unroll
    for (int j = 0; j < 6; ++j) {
      const int idx = g * 6 + j;       // 0..11
      const int mat = idx >> 2;        // 0=Q 1=K 2=V
      const int nt  = idx & 3;
      const bf16* WT = (mat == 0) ? WqT : (mat == 1) ? WkT : WvT;
      v16bf bfr = frag_b16(WT + ((size_t)h * DK + nt * 16) * D_MODEL + k0, D_MODEL);
      acc[j] = wmma_bf16(a, bfr, acc[j]);
    }
    __syncthreads();
  }

  const int lane = threadIdx.x & 31;
  const int n  = lane & 15;
  const int mh = (lane >> 4) << 3;
#pragma unroll
  for (int j = 0; j < 6; ++j) {
    const int idx = g * 6 + j;
    const int mat = idx >> 2;
    const int nt  = idx & 3;
    const float bb = (((mat == 0) ? bq : (mat == 1) ? bk : bv) + h * DK)[nt * 16 + n];
    if (mat == 2) {                     // VT [B,H,64,S]: row dv, contiguous s
      bf16* o = VT + ((size_t)(bh * DK + nt * 16 + n)) * S_LEN + s0 + mt * 16 + mh;
      v8bf vv;
#pragma unroll
      for (int i = 0; i < 8; ++i) vv[i] = (bf16)(acc[j][i] + bb);
      *(v8bf*)o = vv;
    } else {                            // Q/K [B,H,S,64]
      bf16* o = ((mat == 0) ? Q : K) + ((size_t)bh * S_LEN + s0 + mt * 16) * DK;
#pragma unroll
      for (int i = 0; i < 8; ++i)
        o[(size_t)(mh + i) * DK + nt * 16 + n] = (bf16)(acc[j][i] + bb);
    }
  }
}

// ---------------------------------------------------------------------------
// Kernel 2: scores + softmax + attn@V. grid(S/64, B*H).
// 64x1024 f32 score tile in LDS (256KB of 320KB/WGP). attn written exactly
// once with nontemporal stores.
// ---------------------------------------------------------------------------
__global__ __launch_bounds__(256)
void attn_kernel(const bf16* __restrict__ Q, const bf16* __restrict__ K,
                 const bf16* __restrict__ VT, const unsigned char* __restrict__ mask,
                 float* __restrict__ attn, bf16* __restrict__ AO) {
  __shared__ float s_sc[64 * S_LEN];    // 256 KB

  const int s0 = blockIdx.x * 64;
  const int bh = blockIdx.y;
  const int b  = bh >> 4;
  const int wave = threadIdx.x >> 5;
  const int lane = threadIdx.x & 31;
  const int n  = lane & 15;
  const int mh = (lane >> 4) << 3;

  // ---- Phase A: scores = (Q K^T)/8, masked, into LDS ----------------------
  const int mt = wave & 3;
  const int ch = wave >> 2;
  const bf16* qb = Q + ((size_t)bh * S_LEN + s0 + mt * 16) * DK;
  v16bf a0 = frag_a16(qb, DK);
  v16bf a1 = frag_a16(qb + 32, DK);
  const unsigned char* mk = mask + b * S_LEN;

  for (int nt = ch * 32; nt < ch * 32 + 32; ++nt) {
    const int t0 = nt * 16;
    const bf16* kb = K + ((size_t)bh * S_LEN + t0) * DK;
    __builtin_prefetch(kb + 16 * DK, 0, 1);          // next K tile
    v8f c = (v8f){};
    c = wmma_bf16(a0, frag_b16(kb, DK), c);          // K[t][dk]: contiguous dk
    c = wmma_bf16(a1, frag_b16(kb + 32, DK), c);
    const bool ok = mk[t0 + n] != 0;
    float* dst = s_sc + (size_t)(mt * 16 + mh) * S_LEN + t0 + n;
#pragma unroll
    for (int i = 0; i < 8; ++i)
      dst[(size_t)i * S_LEN] = ok ? c[i] * 0.125f : -1e30f;
  }
  __syncthreads();

  // ---- Phase B: fp32 row softmax, vectorized; single NT attn write --------
  for (int rr = 0; rr < 8; ++rr) {
    const int row = wave * 8 + rr;
    v4f* rp4 = (v4f*)(s_sc + (size_t)row * S_LEN);
    float mx = -1e30f;
#pragma unroll
    for (int c4 = 0; c4 < 8; ++c4) {
      v4f v = rp4[c4 * 32 + lane];
      mx = fmaxf(mx, fmaxf(fmaxf(v.x, v.y), fmaxf(v.z, v.w)));
    }
#pragma unroll
    for (int off = 16; off; off >>= 1) mx = fmaxf(mx, __shfl_xor(mx, off));
    float sum = 0.f;
#pragma unroll
    for (int c4 = 0; c4 < 8; ++c4) {
      v4f v = rp4[c4 * 32 + lane];
      v.x = __expf(v.x - mx); v.y = __expf(v.y - mx);
      v.z = __expf(v.z - mx); v.w = __expf(v.w - mx);
      rp4[c4 * 32 + lane] = v;
      sum += v.x + v.y + v.z + v.w;
    }
#pragma unroll
    for (int off = 16; off; off >>= 1) sum += __shfl_xor(sum, off);
    const float inv = 1.f / sum;
    v4f* gp4 = (v4f*)(attn + ((size_t)bh * S_LEN + s0 + row) * S_LEN);
#pragma unroll
    for (int c4 = 0; c4 < 8; ++c4) {
      v4f v = rp4[c4 * 32 + lane];
      v.x *= inv; v.y *= inv; v.z *= inv; v.w *= inv;
      rp4[c4 * 32 + lane] = v;
      __builtin_nontemporal_store(v, gp4 + c4 * 32 + lane);  // never re-read
    }
  }
  __syncthreads();

  // ---- Phase C: out = attn @ V (VT K-contiguous), bf16 to workspace -------
  const int p = wave >> 2;
  const int nt0 = p * 2, nt1 = p * 2 + 1;
  v8f o0 = (v8f){}, o1 = (v8f){};
  for (int k0 = 0; k0 < S_LEN; k0 += 32) {
    v16bf a = frag_a32(s_sc + (size_t)(mt * 16) * S_LEN + k0, S_LEN);
    const bf16* vb = VT + (size_t)bh * DK * S_LEN + k0;     // VT[dv][t]
    o0 = wmma_bf16(a, frag_b16(vb + (size_t)(nt0 * 16) * S_LEN, S_LEN), o0);
    o1 = wmma_bf16(a, frag_b16(vb + (size_t)(nt1 * 16) * S_LEN, S_LEN), o1);
  }
  bf16* ob = AO + ((size_t)bh * S_LEN + s0 + mt * 16) * DK;
#pragma unroll
  for (int i = 0; i < 8; ++i) {
    ob[(size_t)(mh + i) * DK + nt0 * 16 + n] = (bf16)o0[i];
    ob[(size_t)(mh + i) * DK + nt1 * 16 + n] = (bf16)o1[i];
  }
}

// ---------------------------------------------------------------------------
// Kernel 3: projected = multi @ Wo + bo (WoT K-contiguous bf16).
// ---------------------------------------------------------------------------
__global__ __launch_bounds__(256)
void proj_kernel(const bf16* __restrict__ AO, const bf16* __restrict__ WoT,
                 const float* __restrict__ bo, float* __restrict__ out) {
  const int bs = blockIdx.x;
  const int b  = bs >> 4;
  const int s0 = (bs & 15) * 64;
  const int n0 = blockIdx.y * 64;
  const int wave = threadIdx.x >> 5;
  const int lane = threadIdx.x & 31;
  const int mt = wave & 3;
  const int p  = wave >> 2;
  const int nc0 = n0 + p * 32;
  const int nc1 = nc0 + 16;

  v8f o0 = (v8f){}, o1 = (v8f){};
  for (int k0 = 0; k0 < NH * DK; k0 += 32) {
    const int h  = k0 >> 6;
    const int kv = k0 & 63;
    v16bf a = frag_a16(AO + ((size_t)(b * NH + h) * S_LEN + s0 + mt * 16) * DK + kv, DK);
    const bf16* wb = WoT + (size_t)k0;                      // WoT[n][k]
    __builtin_prefetch(wb + (size_t)nc0 * D_MODEL + k0 + 32, 0, 1);
    o0 = wmma_bf16(a, frag_b16(wb + (size_t)nc0 * D_MODEL, D_MODEL), o0);
    o1 = wmma_bf16(a, frag_b16(wb + (size_t)nc1 * D_MODEL, D_MODEL), o1);
  }

  const int n  = lane & 15;
  const int mh = (lane >> 4) << 3;
  const float b0 = bo[nc0 + n];
  const float b1 = bo[nc1 + n];
  float* ob = out + ((size_t)(b * S_LEN + s0 + mt * 16)) * D_MODEL;
#pragma unroll
  for (int i = 0; i < 8; ++i) {
    __builtin_nontemporal_store(o0[i] + b0, ob + (size_t)(mh + i) * D_MODEL + nc0 + n);
    __builtin_nontemporal_store(o1[i] + b1, ob + (size_t)(mh + i) * D_MODEL + nc1 + n);
  }
}

// ---------------------------------------------------------------------------
extern "C" void kernel_launch(void* const* d_in, const int* in_sizes, int n_in,
                              void* d_out, int out_size, void* d_ws, size_t ws_size,
                              hipStream_t stream) {
  (void)in_sizes; (void)n_in; (void)out_size; (void)ws_size;
  const float*         x    = (const float*)d_in[0];
  const unsigned char* mask = (const unsigned char*)d_in[1];
  const float* Wq = (const float*)d_in[2];
  const float* bq = (const float*)d_in[3];
  const float* Wk = (const float*)d_in[4];
  const float* bk = (const float*)d_in[5];
  const float* Wv = (const float*)d_in[6];
  const float* bv = (const float*)d_in[7];
  const float* Wo = (const float*)d_in[8];
  const float* bo = (const float*)d_in[9];

  float* out  = (float*)d_out;                         // [B,S,D_MODEL]
  float* attn = out + (size_t)4 * S_LEN * D_MODEL;     // [B,H,S,S]

  // Workspace (bf16): Q,K,VT,AO (4 x 4.19M) + WqT,WkT,WvT,WoT (4 x 1.05M) = 42MB
  const size_t QN = (size_t)4 * NH * S_LEN * DK;
  const size_t WN = (size_t)NH * D_MODEL * DK;         // == 1024*1024
  bf16* Q   = (bf16*)d_ws;
  bf16* K   = Q + QN;
  bf16* VT  = K + QN;
  bf16* AO  = VT + QN;
  bf16* WqT = AO + QN;
  bf16* WkT = WqT + WN;
  bf16* WvT = WkT + WN;
  bf16* WoT = WvT + WN;

  dim3 blk(256);
  // One-time weight transpose+convert (L2-resident afterwards).
  tcvt_kernel<<<dim3(NH, D_MODEL / 64, DK / 64),      blk, 0, stream>>>(Wq, WqT, D_MODEL, DK);
  tcvt_kernel<<<dim3(NH, D_MODEL / 64, DK / 64),      blk, 0, stream>>>(Wk, WkT, D_MODEL, DK);
  tcvt_kernel<<<dim3(NH, D_MODEL / 64, DK / 64),      blk, 0, stream>>>(Wv, WvT, D_MODEL, DK);
  tcvt_kernel<<<dim3(1, D_MODEL / 64, D_MODEL / 64),  blk, 0, stream>>>(Wo, WoT, D_MODEL, D_MODEL);

  qkv_kernel <<<dim3(4 * NH, S_LEN / 64), blk, 0, stream>>>(x, WqT, bq, WkT, bk, WvT, bv, Q, K, VT);
  attn_kernel<<<dim3(S_LEN / 64, 4 * NH), blk, 0, stream>>>(Q, K, VT, mask, attn, AO);
  proj_kernel<<<dim3(4 * S_LEN / 64, D_MODEL / 64), blk, 0, stream>>>(AO, WoT, bo, out);
}
